// VectorQuantizer2_45595372814697
// MI455X (gfx1250) — matrix operand order repaired and checked by reading, not compiled, not run on hardware
//
#include <hip/hip_runtime.h>

// ---------------------------------------------------------------------------
// VQ-VAE VectorQuantizer for MI455X (gfx1250, wave32, WMMA).
//   z:   [8][256][64][64] f32   -> N=32768 tokens of dim 256
//   emb: [8192][256] f32
// Outputs (concat in d_out, f32): z_q [8][256][64][64], loss [1], indices [32768]
//
// Main kernel: per-block LDS-shared, double-buffered codebook tiles feeding
// chained v_wmma_f32_16x16x32_bf16; global prefetch of tile t+1 overlaps the
// WMMA chain of tile t.
// ---------------------------------------------------------------------------

typedef __attribute__((ext_vector_type(16))) __bf16 v16bf;
typedef __attribute__((ext_vector_type(8)))  __bf16 v8bf;
typedef __attribute__((ext_vector_type(8)))  float  v8f;

#define NE    8192
#define EDIM  256
#define NTOK  32768   // 8 * 64 * 64
#define NELEM 8388608 // 8 * 256 * 64 * 64
#define BETA  0.25f
#define LDS_STRIDE 264   // 132 dwords/row; 132 % 64 == 4 -> conflict-minimal

// ---------------------------------------------------------------------------
// Kernel 1: transpose z (b,c,h,w) f32 -> zbf (n, c) bf16, n = b*4096 + h*64 + w
// ---------------------------------------------------------------------------
__global__ __launch_bounds__(256)
void vq_transpose_kernel(const float* __restrict__ z, __bf16* __restrict__ zbf)
{
    __shared__ float tile[32][33];
    const int b  = blockIdx.z;
    const int p0 = blockIdx.x * 32;   // spatial position h*64+w
    const int c0 = blockIdx.y * 32;   // channel
    const int tx = threadIdx.x & 31;
    const int ty = threadIdx.x >> 5;  // 0..7

    const float* src = z + ((size_t)b * 256 + c0) * 4096 + p0;
#pragma unroll
    for (int i = 0; i < 32; i += 8)
        tile[ty + i][tx] = src[(size_t)(ty + i) * 4096 + tx];
    __syncthreads();

    __bf16* dst = zbf + ((size_t)b * 4096 + p0) * 256 + c0;
#pragma unroll
    for (int i = 0; i < 32; i += 8)
        dst[(size_t)(ty + i) * 256 + tx] = (__bf16)tile[tx][ty + i];
}

// ---------------------------------------------------------------------------
// Kernel 2: embedding f32 -> bf16, plus ||e_k||^2.  grid: 8192 blocks x 256.
// ---------------------------------------------------------------------------
__global__ __launch_bounds__(256)
void vq_embed_prep_kernel(const float* __restrict__ emb,
                          __bf16* __restrict__ ebf,
                          float* __restrict__ enorm)
{
    const int k = blockIdx.x;
    const int c = threadIdx.x;
    const float v = emb[(size_t)k * EDIM + c];
    ebf[(size_t)k * EDIM + c] = (__bf16)v;

    __shared__ float red[256];
    red[c] = v * v;
    __syncthreads();
    for (int s = 128; s > 0; s >>= 1) {
        if (c < s) red[c] += red[c + s];
        __syncthreads();
    }
    if (c == 0) enorm[k] = red[0];
}

// ---------------------------------------------------------------------------
// Kernel 3: distance GEMM + argmin.
//   score[m][n] = ||e_n||^2 - 2 * (z_m . e_n)    (||z||^2 constant per row)
// Block = 256 threads = 8 waves; wave w owns rows mBase + w*16 .. +15.
// All waves share the 16-code B tile staged in LDS (double-buffered); tile
// t+1 is prefetched from global while tile t runs 8 chained bf16 WMMAs.
// grid: 256 blocks -> 128 rows/block.
// ---------------------------------------------------------------------------
__global__ __launch_bounds__(256)
void vq_argmin_kernel(const __bf16* __restrict__ zbf,
                      const __bf16* __restrict__ ebf,
                      const float* __restrict__ enorm,
                      int*   __restrict__ idx_out,
                      float* __restrict__ idxf_out)
{
    __shared__ __bf16 bstage[2][16 * LDS_STRIDE];

    const int tid   = threadIdx.x;
    const int lane  = tid & 31;
    const int wave  = tid >> 5;
    const int mBase = (blockIdx.x * 8 + wave) * 16;
    const int col   = lane & 15;   // A row / B column / D column
    const int hv    = lane >> 4;   // lane-half selector

    // ---- staging coordinates: thread stages 32B of the 8KB tile ----
    const int sr = tid >> 4;           // tile row   0..15
    const int sk = (tid & 15) * 16;    // tile k off 0..240
    const __bf16* gsrc = ebf + (size_t)sr * EDIM + sk;   // + t*4096 per tile
    __bf16* ldst0 = &bstage[0][sr * LDS_STRIDE + sk];
    __bf16* ldst1 = &bstage[1][sr * LDS_STRIDE + sk];

    // ---- Preload A tile (16 rows x 256 K) in ISA 16-bit A layout ----
    // lanes 0-15: elems 0-7 <- K=k0+0..7,   elems 8-15 <- K=k0+16..23
    // lanes16-31: elems 0-7 <- K=k0+8..15,  elems 8-15 <- K=k0+24..31
    v16bf A[8];
    const __bf16* zrow = zbf + (size_t)(mBase + col) * EDIM;
#pragma unroll
    for (int ks = 0; ks < 8; ++ks) {
        const int k0 = ks * 32;
        v8bf c0 = *(const v8bf*)(zrow + k0 + hv * 8);
        v8bf c1 = *(const v8bf*)(zrow + k0 + 16 + hv * 8);
#pragma unroll
        for (int i = 0; i < 8; ++i) { A[ks][i] = c0[i]; A[ks][i + 8] = c1[i]; }
    }

    // ---- prologue: stage tile 0 into buffer 0 ----
    {
        v8bf s0 = *(const v8bf*)(gsrc);
        v8bf s1 = *(const v8bf*)(gsrc + 8);
        *(v8bf*)(ldst0)     = s0;
        *(v8bf*)(ldst0 + 8) = s1;
    }
    __syncthreads();

    float runMin[8];
    int   runIdx[8];
#pragma unroll
    for (int r = 0; r < 8; ++r) { runMin[r] = 3.4e38f; runIdx[r] = 0; }

    for (int t = 0; t < NE / 16; ++t) {
        const int nb = t * 16;

        // prefetch tile t+1 (last iter over-reads 8KB into adjacent ws; unused)
        v8bf n0 = *(const v8bf*)(gsrc + (size_t)(t + 1) * 4096);
        v8bf n1 = *(const v8bf*)(gsrc + (size_t)(t + 1) * 4096 + 8);

        // compute on tile t from LDS buffer (t & 1)
        const __bf16* brow = &bstage[t & 1][col * LDS_STRIDE + hv * 8];
        v8f acc = {};
#pragma unroll
        for (int ks = 0; ks < 8; ++ks) {
            v8bf b0 = *(const v8bf*)(brow + ks * 32);
            v8bf b1 = *(const v8bf*)(brow + ks * 32 + 16);
            v16bf B;
#pragma unroll
            for (int i = 0; i < 8; ++i) { B[i] = b0[i]; B[i + 8] = b1[i]; }
            acc = __builtin_amdgcn_wmma_f32_16x16x32_bf16(
                      false, A[ks], false, B, (short)0, acc, false, false);
        }
        const float en = enorm[nb + col];
#pragma unroll
        for (int r = 0; r < 8; ++r) {
            // D layout: VGPR r, lane L -> row (r + 8*hv), col (L&15)
            const float s = en - 2.0f * acc[r];
            if (s < runMin[r]) { runMin[r] = s; runIdx[r] = nb + col; }
        }

        // stage tile t+1 into the other buffer; one barrier per iteration
        __bf16* ld = (t & 1) ? ldst0 : ldst1;
        *(v8bf*)(ld)     = n0;
        *(v8bf*)(ld + 8) = n1;
        __syncthreads();
    }

    // ---- argmin across the 16 columns (within each lane-half) per row ----
#pragma unroll
    for (int r = 0; r < 8; ++r) {
        float v  = runMin[r];
        int   ix = runIdx[r];
#pragma unroll
        for (int off = 1; off < 16; off <<= 1) {
            const float ov = __shfl_xor(v, off, 32);
            const int   oi = __shfl_xor(ix, off, 32);
            if (ov < v || (ov == v && oi < ix)) { v = ov; ix = oi; }
        }
        if (col == 0) {
            const int m = mBase + r + hv * 8;
            idx_out[m]  = ix;
            idxf_out[m] = (float)ix;
        }
    }
}

// ---------------------------------------------------------------------------
// Kernel 4: gather z_q (f32, back to b,c,h,w) + per-block loss partials.
// ---------------------------------------------------------------------------
__global__ __launch_bounds__(256)
void vq_gather_loss_kernel(const float* __restrict__ z,
                           const float* __restrict__ emb,
                           const int* __restrict__ idx,
                           float* __restrict__ zq,
                           float* __restrict__ partial)
{
    float acc = 0.0f;
#pragma unroll
    for (int j = 0; j < 4; ++j) {
        const size_t f = (size_t)blockIdx.x * 1024 + (size_t)j * 256 + threadIdx.x;
        const int w = (int)(f & 63);
        const int h = (int)((f >> 6) & 63);
        const int c = (int)((f >> 12) & 255);
        const int b = (int)(f >> 20);
        const int n = b * 4096 + h * 64 + w;
        const int k = idx[n];
        const float e = emb[(size_t)k * EDIM + c];  // 8MB, L2-resident gather
        const float d = e - z[f];
        zq[f] = e;                                  // straight-through value == e
        acc += d * d;
    }
    __shared__ float red[256];
    red[threadIdx.x] = acc;
    __syncthreads();
    for (int s = 128; s > 0; s >>= 1) {
        if (threadIdx.x < s) red[threadIdx.x] += red[threadIdx.x + s];
        __syncthreads();
    }
    if (threadIdx.x == 0) partial[blockIdx.x] = red[0];
}

// ---------------------------------------------------------------------------
// Kernel 5: deterministic final reduction -> loss = (1+beta)*mean(diff^2)
// ---------------------------------------------------------------------------
__global__ __launch_bounds__(256)
void vq_loss_final_kernel(const float* __restrict__ partial, float* __restrict__ loss_out)
{
    __shared__ float red[256];
    float a = 0.0f;
    for (int i = threadIdx.x; i < 8192; i += 256) a += partial[i];
    red[threadIdx.x] = a;
    __syncthreads();
    for (int s = 128; s > 0; s >>= 1) {
        if (threadIdx.x < s) red[threadIdx.x] += red[threadIdx.x + s];
        __syncthreads();
    }
    if (threadIdx.x == 0)
        loss_out[0] = red[0] * (1.0f + BETA) / (float)NELEM;
}

// ---------------------------------------------------------------------------
extern "C" void kernel_launch(void* const* d_in, const int* in_sizes, int n_in,
                              void* d_out, int out_size, void* d_ws, size_t ws_size,
                              hipStream_t stream)
{
    const float* z   = (const float*)d_in[0];   // 8*256*64*64
    const float* emb = (const float*)d_in[1];   // 8192*256
    float* out = (float*)d_out;

    // workspace layout (bytes)
    char* ws = (char*)d_ws;
    __bf16* ebf    = (__bf16*)(ws);                              //  4 MB
    __bf16* zbf    = (__bf16*)(ws + (size_t)NE * EDIM * 2);      // 16 MB (also absorbs prefetch over-read)
    int*    idx    = (int*)   (ws + (size_t)NE * EDIM * 2
                                  + (size_t)NTOK * EDIM * 2);    // 128 KB
    float*  enorm  = (float*) ((char*)idx + (size_t)NTOK * 4);   //  32 KB
    float*  part   = (float*) ((char*)enorm + (size_t)NE * 4);   //  32 KB

    float* zq_out   = out;               // [0, 8388608)
    float* loss_out = out + NELEM;       // [8388608]
    float* idxf_out = out + NELEM + 1;   // [8388609, +32768)

    // 1) z -> bf16 token-major
    vq_transpose_kernel<<<dim3(4096 / 32, 256 / 32, 8), 256, 0, stream>>>(z, zbf);
    // 2) embedding -> bf16 + norms
    vq_embed_prep_kernel<<<NE, 256, 0, stream>>>(emb, ebf, enorm);
    // 3) WMMA distance GEMM + argmin (LDS double-buffered codebook tiles)
    vq_argmin_kernel<<<NTOK / 128, 256, 0, stream>>>(zbf, ebf, enorm, idx, idxf_out);
    // 4) gather + loss partials
    vq_gather_loss_kernel<<<8192, 256, 0, stream>>>(z, emb, idx, zq_out, part);
    // 5) final loss
    vq_loss_final_kernel<<<1, 256, 0, stream>>>(part, loss_out);
}
